// PitchShift_86148454023388
// MI455X (gfx1250) — compile-verified, matrix operand order checked by reading, not run on hardware
//
#include <hip/hip_runtime.h>
#include <math.h>

// ---------------- constants ----------------
#define SRATE    16000.0f
#define FLOORF0  60.0f
#define EPSF     1e-12f
#define MAXSEG   256
#define PKSTRIDE 4096
#define MAXPK    (MAXSEG * PKSTRIDE)
#define HBINS    4096          // f0 histogram, 1/16 Hz bins over [0,256)
#define REGION   512           // LDS staging for lag windows (16B-aligned base)
#define TPAD     304           // LDS staging for template (9 chunks of K=32 + align slack)
#define MAXR     144           // max lags per step (9 groups of 16)

typedef __attribute__((ext_vector_type(16))) _Float16 v16h;
typedef __attribute__((ext_vector_type(8)))  float    v8f;

// ---------------- small helpers ----------------
__device__ inline int iclamp(int v, int lo, int hi) { return v < lo ? lo : (v > hi ? hi : v); }
__device__ inline int imin2(int a, int b) { return a < b ? a : b; }
__device__ inline int imax2(int a, int b) { return a > b ? a : b; }

__device__ inline float waveSum(float v) {
  for (int m = 16; m > 0; m >>= 1) v += __shfl_xor(v, m, 32);
  return v;
}
__device__ inline float waveMax(float v) {
  for (int m = 16; m > 0; m >>= 1) v = fmaxf(v, __shfl_xor(v, m, 32));
  return v;
}

// linear interp of pitch frames to sample resolution (matches _interp_linear)
__device__ inline float f0_at(const float* __restrict__ p, int S, int T, int t) {
  double pos = ((double)t + 0.5) * ((double)S / (double)T) - 0.5;
  double mx = (double)(S - 1);
  pos = pos < 0.0 ? 0.0 : (pos > mx ? mx : pos);
  int lo = (int)pos;                    // pos >= 0 so trunc == floor
  int hi = lo + 1; if (hi > S - 1) hi = S - 1;
  float frac = (float)(pos - (double)lo);
  return p[lo] * (1.0f - frac) + p[hi] * frac;
}

__device__ inline float f0_shifted(const float* __restrict__ p, int S, int T, int t,
                                   float shift, float median, float range) {
  float v = f0_at(p, S, T, t) * shift;
  return (v > 0.0f) ? (median + (v - median) * range) : 0.0f;
}

__device__ inline int lowerBound(const int* __restrict__ a, int n, int key) {
  int lo = 0, hi = n;
  while (lo < hi) { int mid = (lo + hi) >> 1; if (a[mid] < key) lo = mid + 1; else hi = mid; }
  return lo;
}

__device__ inline void isort(int* a, int n) {
  for (int k = 1; k < n; ++k) {
    int v = a[k]; int j = k - 1;
    while (j >= 0 && a[j] > v) { a[j + 1] = a[j]; --j; }
    a[j + 1] = v;
  }
}

// ---------------- kernel 0: init ----------------
__global__ void initKernel(float* __restrict__ out, int T, unsigned* __restrict__ hist, int HB,
                           float* sumAccum, unsigned* gpeakBits, unsigned* f0minBits, int* cnts,
                           int* nSegPtr, int* nPeaksPtr, int* segPeakCount, int MS) {
  long long idx0 = (long long)blockIdx.x * blockDim.x + threadIdx.x;
  long long stride = (long long)gridDim.x * blockDim.x;
  for (long long t = idx0; t < T; t += stride) out[t] = 0.0f;
  for (long long t = idx0; t < HB; t += stride) hist[t] = 0u;
  for (long long t = idx0; t < MS; t += stride) segPeakCount[t] = 0;
  if (idx0 == 0) {
    *sumAccum = 0.0f; *gpeakBits = 0u; *f0minBits = 0x7F800000u; // +inf bits
    cnts[0] = 0; cnts[1] = 0; *nSegPtr = 0; *nPeaksPtr = 0;
  }
}

// ---------------- kernel 1/2: mean ----------------
__global__ void sumKernel(const float* __restrict__ snd, float* sumAccum, int T) {
  __shared__ float sb[256];
  float acc = 0.0f;
  for (long long t = (long long)blockIdx.x * blockDim.x + threadIdx.x; t < T;
       t += (long long)gridDim.x * blockDim.x)
    acc += snd[t];
  sb[threadIdx.x] = acc; __syncthreads();
  for (int s = 128; s > 0; s >>= 1) {
    if ((int)threadIdx.x < s) sb[threadIdx.x] += sb[threadIdx.x + s];
    __syncthreads();
  }
  if (threadIdx.x == 0) atomicAdd(sumAccum, sb[0]);
}

__global__ void meanKernel(const float* sumAccum, float* meanVal, int T) {
  if (threadIdx.x == 0 && blockIdx.x == 0) *meanVal = *sumAccum / (float)T;
}

// ---------------- kernel 3: f16 convert, gpeak, f0 histogram, voiced runs ----------------
__global__ void prepKernel(const float* __restrict__ snd, const float* __restrict__ pitch,
                           const float* __restrict__ meanVal, _Float16* __restrict__ xh,
                           unsigned* __restrict__ gpeakBits, unsigned* __restrict__ f0minBits,
                           unsigned* __restrict__ hist, int* __restrict__ segStartRaw,
                           int* __restrict__ segEndRaw, int* __restrict__ cnts, int S, int T) {
  __shared__ unsigned lh[HBINS];
  for (int j = threadIdx.x; j < HBINS; j += blockDim.x) lh[j] = 0u;
  __syncthreads();
  float mean = *meanVal;
  float lmax = 0.0f, lminf = 3.4e38f;
  for (long long t = (long long)blockIdx.x * blockDim.x + threadIdx.x; t < T;
       t += (long long)gridDim.x * blockDim.x) {
    float x = snd[t] - mean;
    xh[t] = (_Float16)x;
    float ax = fabsf(x); if (ax > lmax) lmax = ax;
    float f = f0_at(pitch, S, T, (int)t);
    bool v  = f > 0.0f;
    bool vp = (t > 0) ? (f0_at(pitch, S, T, (int)t - 1) > 0.0f) : false;
    if (v) {
      int b = (int)(f * 16.0f); b = iclamp(b, 0, HBINS - 1);
      atomicAdd(&lh[b], 1u);
      if (f < lminf) lminf = f;
    }
    if (v && !vp) { int k = atomicAdd(&cnts[0], 1); if (k < MAXSEG) segStartRaw[k] = (int)t; }
    if (!v && vp) { int k = atomicAdd(&cnts[1], 1); if (k < MAXSEG) segEndRaw[k] = (int)t; }
    if (v && t == (long long)T - 1) {
      int k = atomicAdd(&cnts[1], 1); if (k < MAXSEG) segEndRaw[k] = T;
    }
  }
  atomicMax(gpeakBits, __float_as_uint(lmax));
  atomicMin(f0minBits, __float_as_uint(lminf));
  __syncthreads();
  for (int j = threadIdx.x; j < HBINS; j += blockDim.x)
    if (lh[j]) atomicAdd(&hist[j], lh[j]);
}

// ---------------- kernel 4: sort segments, median, tail ----------------
__global__ void segSortKernel(const int* __restrict__ cnts, int* segStartRaw, int* segEndRaw,
                              int* segStart, int* segEnd, int* nSegPtr, int* tailPtr,
                              const unsigned* __restrict__ hist, float* medianVal,
                              const float* __restrict__ pshiftP, int T) {
  if (threadIdx.x != 0 || blockIdx.x != 0) return;
  int nsRaw = imin2(cnts[0], MAXSEG);
  int neRaw = imin2(cnts[1], MAXSEG);
  isort(segStartRaw, nsRaw);
  isort(segEndRaw, neRaw);
  int ns = imin2(nsRaw, neRaw);
  for (int k = 0; k < ns; ++k) { segStart[k] = segStartRaw[k]; segEnd[k] = segEndRaw[k]; }
  *nSegPtr = ns;
  *tailPtr = (ns > 0) ? imin2(segEnd[ns - 1], T) : 0;
  unsigned long long total = 0;
  for (int b = 0; b < HBINS; ++b) total += hist[b];
  float medianRaw = 0.0f;
  if (total > 0) {
    unsigned long long target = (total - 1) / 2, cum = 0;
    for (int b = 0; b < HBINS; ++b) {
      cum += hist[b];
      if (cum > target) { medianRaw = ((float)b + 0.5f) / 16.0f; break; }
    }
  }
  *medianVal = medianRaw * (*pshiftP);
}

// ---------------- WMMA-based normalized cross-correlation step ----------------
struct PeakRes { int w; float corr; int i; float peak; };

__device__ PeakRes find_peak(const _Float16* __restrict__ xh, const float* __restrict__ pitch,
                             int S, int T, int i, int dirRight,
                             _Float16* lsig, _Float16* ltmp, float* lcorr, float* lpre,
                             float* sred, int* sredi) {
  const int lane = threadIdx.x;
  PeakRes res;
  float f = f0_at(pitch, S, T, i);
  int w = (int)(SRATE / fmaxf(f, FLOORF0));
  if (w < 1) w = 1;
  int s = imax2(i - w / 2, 0);
  int cl, cr;
  if (!dirRight) {
    cl = imax2((int)((float)i - 1.75f * (float)w), 0);
    cr = imax2((int)((float)i - 1.3f  * (float)w), 0);
  } else {
    cl = (int)((float)i + 0.3f  * (float)w);
    cr = (int)((float)i + 0.75f * (float)w);
  }
  res.w = w;
  if (cl == cr || (T - cl) < w || (s + w) > T) {
    res.corr = -1.0f; res.i = i; res.peak = 0.0f; return res;
  }
  int hiEnd = imin2(cr + w, T);
  int R = hiEnd - cl - w + 1;
  if (R < 1) { res.corr = -1.0f; res.i = i; res.peak = 0.0f; return res; }
  if (R > MAXR) R = MAXR;

  // 16B-aligned staging bases (slack values before cl / s are harmless:
  // template is zero-padded, so A-side padding multiplies by zero, and
  // out-of-range lag rows are never read back from the accumulator).
  int clA = cl & ~7;  int off0 = cl - clA;   // 0..7
  int sA  = s  & ~7;  int toff = s  - sA;    // 0..7

  if (lane == 0) __builtin_prefetch(xh + clA, 0, 1);   // gfx1250 global_prefetch_b8

  // Stage candidate region with b128 loads where fully in-range
  {
    const uint4* vsrc = (const uint4*)(xh + clA);
    for (int q = lane; q < REGION / 8; q += 32) {
      int j0 = q * 8;
      int g0 = clA + j0;
      if (g0 + 8 <= T) {
        *(uint4*)(lsig + j0) = vsrc[q];
      } else {
        for (int e = 0; e < 8; ++e) {
          int g = g0 + e;
          lsig[j0 + e] = (g < T) ? xh[g] : (_Float16)0.0f;
        }
      }
    }
  }
  // Stage template (zero-padded beyond w: REQUIRED for B operand)
  {
    const uint4* tsrc = (const uint4*)(xh + sA);
    int tvalid = toff + w;           // ltmp[j] valid for j in [toff, tvalid)
    for (int q = lane; q < TPAD / 8; q += 32) {
      int j0 = q * 8;
      if (j0 + 8 <= tvalid && sA + j0 + 8 <= T) {
        *(uint4*)(ltmp + j0) = tsrc[q];
      } else {
        for (int e = 0; e < 8; ++e) {
          int j = j0 + e;
          int g = sA + j;
          ltmp[j] = (j < tvalid && g < T) ? xh[g] : (_Float16)0.0f;
        }
      }
    }
  }
  __syncthreads();

  // template norm
  float tn = 0.0f;
  for (int j = lane; j < w; j += 32) { float v = (float)ltmp[toff + j]; tn += v * v; }
  tn = fmaxf(sqrtf(waveSum(tn)), EPSF);

  // exclusive prefix sum of squares over the whole staged region:
  // lpre[j] = sum_{k<j} lsig[k]^2  ->  norm^2(r) = lpre[off0+r+w] - lpre[off0+r]
  {
    int base = lane * (REGION / 32);           // 16 contiguous elems per lane
    float loc[REGION / 32];
    float acc = 0.0f;
    #pragma unroll
    for (int j = 0; j < REGION / 32; ++j) {
      float v = (float)lsig[base + j];
      acc += v * v;
      loc[j] = acc;                            // inclusive within chunk
    }
    float sc = acc;                            // wave inclusive scan of chunk totals
    for (int d = 1; d < 32; d <<= 1) {
      float o = __shfl_up(sc, d, 32);
      if (lane >= d) sc += o;
    }
    float chunkExcl = sc - acc;
    if (lane == 0) lpre[0] = 0.0f;
    #pragma unroll
    for (int j = 0; j < REGION / 32; ++j) lpre[base + j + 1] = chunkExcl + loc[j];
  }
  __syncthreads();

  // sliding dot products: K-chunk outer loop so the template tile (B) is
  // loaded from LDS once per chunk; 9 persistent f32 accumulators.
  int nk   = (w + 31) >> 5;
  int ngrp = (R + 15) >> 4;
  int row  = lane & 15;
  int kb   = (lane >> 4) << 4;     // K sub-half handled by this lane (A and B layouts)
  v8f accs[9];
  {
    v8f zero = {};
    #pragma unroll
    for (int g = 0; g < 9; ++g) accs[g] = zero;
  }
  for (int c = 0; c < nk; ++c) {
    v16h b;
    int tb = toff + c * 32 + kb;
    #pragma unroll
    for (int j = 0; j < 16; ++j) b[j] = ltmp[tb + j];
    #pragma unroll
    for (int g = 0; g < 9; ++g) {
      if (g < ngrp) {
        v16h a;
        int ab = off0 + g * 16 + row + kb + c * 32;
        #pragma unroll
        for (int j = 0; j < 16; ++j) a[j] = lsig[ab + j];
        accs[g] = __builtin_amdgcn_wmma_f32_16x16x32_f16(false, a, false, b, (short)0,
                                                         accs[g], false, false);
      }
    }
  }
  // D: lane L holds rows (L>>4)*8 + j at column L&15; all columns identical -> take col 0
  if ((lane & 15) == 0) {
    int rbl = (lane >> 4) << 3;
    #pragma unroll
    for (int g = 0; g < 9; ++g) {
      if (g < ngrp) {
        #pragma unroll
        for (int j = 0; j < 8; ++j) lcorr[g * 16 + rbl + j] = accs[g][j];
      }
    }
  }
  __syncthreads();

  // per-lag normalization + argmax (norms from prefix sums, O(R) total)
  float bv = -1e30f; int bi = 0;
  for (int r = lane; r < R; r += 32) {
    float nn = fmaxf(lpre[off0 + r + w] - lpre[off0 + r], 0.0f);
    float cc = lcorr[r] / (fmaxf(sqrtf(nn), EPSF) * tn);
    if (cc > bv) { bv = cc; bi = r; }
  }
  sred[lane] = bv; sredi[lane] = bi; __syncthreads();
  if (lane == 0) {
    float best = sred[0]; int bidx = sredi[0];
    for (int l = 1; l < 32; ++l)
      if (sred[l] > best || (sred[l] == best && sredi[l] < bidx)) { best = sred[l]; bidx = sredi[l]; }
    sred[0] = best; sredi[0] = bidx;
  }
  __syncthreads();
  float corr = sred[0]; int r = sredi[0];
  __syncthreads();

  float pk = 0.0f;
  for (int j = lane; j < w; j += 32) pk = fmaxf(pk, fabsf((float)lsig[off0 + r + j]));
  pk = waveMax(pk);

  res.corr = corr; res.i = i + (r + cl) - s; res.peak = pk;
  return res;
}

// ---------------- kernel 5: per-segment peak tracking (1 wave / voiced run) ----------------
__global__ void __launch_bounds__(32) findPeaksKernel(
    const float* __restrict__ snd, const float* __restrict__ pitch,
    const _Float16* __restrict__ xh, const float* __restrict__ meanVal,
    const unsigned* __restrict__ gpeakBits, const int* __restrict__ nSegPtr,
    const int* __restrict__ segStart, const int* __restrict__ segEnd,
    int* __restrict__ pkStage, int* __restrict__ segPeakCount, int S, int T) {
  __shared__ __align__(16) _Float16 lsig[REGION];
  __shared__ __align__(16) _Float16 ltmp[TPAD];
  __shared__ float lcorr[MAXR];
  __shared__ float lpre[REGION + 1];
  __shared__ float sred[32];
  __shared__ int   sredi[32];

  int seg = blockIdx.x;
  if (seg >= *nSegPtr) return;
  int lane = threadIdx.x;
  int left = segStart[seg], right = segEnd[seg];
  if (right <= left) { if (lane == 0) segPeakCount[seg] = 0; return; }
  float mean  = *meanVal;
  float gpeak = __uint_as_float(*gpeakBits);
  int* mypk = pkStage + (size_t)seg * PKSTRIDE;

  // starting extremum near the segment middle
  int middle = (left + right) / 2;
  float fm = fmaxf(f0_at(pitch, S, T, middle), 1.0f);
  int w0 = imax2((int)(SRATE / fm), 1);
  int s0 = imax2(middle - w0 / 2, 0);
  float mn = 1e30f, mx = -1e30f; int imn = 0, imx = 0;
  for (int j = lane; j < w0; j += 32) {
    int t = s0 + j; if (t >= T) break;
    float v = snd[t] - mean;
    if (v < mn) { mn = v; imn = j; }
    if (v > mx) { mx = v; imx = j; }
  }
  sred[lane] = mn; sredi[lane] = imn; __syncthreads();
  if (lane == 0) {
    float b = sred[0]; int bi = sredi[0];
    for (int l = 1; l < 32; ++l)
      if (sred[l] < b || (sred[l] == b && sredi[l] < bi)) { b = sred[l]; bi = sredi[l]; }
    sred[0] = b; sredi[0] = bi;
  }
  __syncthreads(); mn = sred[0]; imn = sredi[0]; __syncthreads();
  sred[lane] = mx; sredi[lane] = imx; __syncthreads();
  if (lane == 0) {
    float b = sred[0]; int bi = sredi[0];
    for (int l = 1; l < 32; ++l)
      if (sred[l] > b || (sred[l] == b && sredi[l] < bi)) { b = sred[l]; bi = sredi[l]; }
    sred[0] = b; sredi[0] = bi;
  }
  __syncthreads(); mx = sred[0]; imx = sredi[0]; __syncthreads();

  int i = (mn == mx) ? middle : (s0 + ((fabsf(mn) > fabsf(mx)) ? imn : imx));
  int backup = i;
  float added_right = -1e30f;
  int n = 0;

  // LEFT scan (appends descending positions)
  for (int it = 0; it < 50000; ++it) {
    PeakRes r = find_peak(xh, pitch, S, T, i, 0, lsig, ltmp, lcorr, lpre, sred, sredi);
    i = r.i;
    if (r.corr == -1.0f) i -= r.w;
    if (i < left) {
      if (r.corr > 0.7f && r.peak > 0.023333f * gpeak &&
          ((float)i - added_right > 0.8f * (float)r.w)) {
        if (lane == 0 && n < PKSTRIDE) mypk[n] = iclamp(i, 0, T - 1);
        n++;
      }
      break;
    }
    if (r.corr > 0.3f && (r.peak == 0.0f || r.peak > 0.01f * gpeak)) {
      if ((float)i - added_right > 0.8f * (float)r.w) {
        if (lane == 0 && n < PKSTRIDE) mypk[n] = iclamp(i, 0, T - 1);
        n++;
      }
    }
  }
  int nLeft = imin2(n, PKSTRIDE);
  __syncthreads();
  for (int j = lane; j < nLeft / 2; j += 32) {   // reverse -> ascending
    int a = mypk[j]; mypk[j] = mypk[nLeft - 1 - j]; mypk[nLeft - 1 - j] = a;
  }
  __syncthreads();

  // RIGHT scan (appends ascending positions)
  i = backup;
  for (int it = 0; it < 50000; ++it) {
    PeakRes r = find_peak(xh, pitch, S, T, i, 1, lsig, ltmp, lcorr, lpre, sred, sredi);
    i = r.i;
    if (r.corr == -1.0f) i += r.w;
    if (i >= right) {
      if (r.corr > 0.7f && r.peak > 0.023333f * gpeak) {
        if (lane == 0 && n < PKSTRIDE) mypk[n] = iclamp(i, 0, T - 1);
        n++; added_right = (float)i;
      }
      break;
    }
    if (r.corr > 0.3f && (r.peak == 0.0f || r.peak > 0.01f * gpeak)) {
      if (lane == 0 && n < PKSTRIDE) mypk[n] = iclamp(i, 0, T - 1);
      n++; added_right = (float)i;
    }
  }
  if (lane == 0) segPeakCount[seg] = imin2(n, PKSTRIDE);
}

// ---------------- kernel 6: compact per-segment peak lists ----------------
__global__ void compactKernel(const int* __restrict__ nSegPtr, const int* __restrict__ segPeakCount,
                              int* __restrict__ segPeakOff, const int* __restrict__ pkStage,
                              int* __restrict__ pkCompact, int* __restrict__ nPeaksPtr) {
  __shared__ int soff[MAXSEG + 1];
  int ns = *nSegPtr;
  if (threadIdx.x == 0) {
    int acc = 0;
    for (int s2 = 0; s2 < ns; ++s2) { soff[s2] = acc; segPeakOff[s2] = acc; acc += segPeakCount[s2]; }
    soff[ns] = acc;
    *nPeaksPtr = imin2(acc, MAXPK);
  }
  __syncthreads();
  for (int s2 = 0; s2 < ns; ++s2) {
    int cnt = segPeakCount[s2];
    int offc = soff[s2];
    for (int j = threadIdx.x; j < cnt; j += blockDim.x) {
      int d = offc + j;
      if (d < MAXPK) pkCompact[d] = pkStage[(size_t)s2 * PKSTRIDE + j];
    }
  }
}

// ---------------- kernel 7: PSOLA plan + overlap-add scatter ----------------
__global__ void psolaKernel(const float* __restrict__ snd, const float* __restrict__ pitch,
                            const float* __restrict__ meanVal, const float* __restrict__ medianVal,
                            const unsigned* __restrict__ f0minBits, const float* __restrict__ pshiftP,
                            const float* __restrict__ prangeP, const int* __restrict__ nSegPtr,
                            const int* __restrict__ segStart, const int* __restrict__ segEnd,
                            const int* __restrict__ pkCompact, const int* __restrict__ nPeaksPtr,
                            float* __restrict__ out, int S, int T) {
  int seg = blockIdx.x;
  int ns = *nSegPtr;
  if (seg >= ns) return;
  float mean = *meanVal, median = *medianVal;
  float shift = *pshiftP, range = *prangeP;
  int np = *nPeaksPtr;
  float f0minRaw = __uint_as_float(*f0minBits);
  float fmin_t = median + (f0minRaw * shift - median) * range;   // transform is monotone
  float max_w = 1.25f * SRATE / fmaxf(fmin_t, 1e-3f);
  int left = segStart[seg], right = segEnd[seg];

  // hann-windowed passthrough of the unvoiced gap before this segment
  int gap0 = (seg == 0) ? 0 : segEnd[seg - 1];
  int gn = left - gap0;
  for (int j = threadIdx.x; j < gn; j += blockDim.x) {
    float wv = 0.5f - 0.5f * __cosf(6.28318530718f * (float)j / (float)gn);
    int t = gap0 + j;
    atomicAdd(&out[t], wv * (snd[t] - mean));
  }
  if (np <= 0) return;

  // grains (control flow is uniform: every thread computes identical scalars)
  int lv = left;
  for (int it = 0; it < 400000 && lv < right; ++it) {
    float fs = f0_shifted(pitch, S, T, lv, shift, median, range);
    int period = (int)(SRATE / fmaxf(fs, FLOORF0));
    int p0 = lowerBound(pkCompact, np, lv);
    int p;
    if (p0 >= np) p = np - 1;
    else if (p0 == 0) p = 0;
    else p = ((lv - pkCompact[p0 - 1]) <= (pkCompact[p0] - lv)) ? (p0 - 1) : p0;
    int pk = pkCompact[p];
    int lw = period / 2, rw = period / 2;
    if (p > 0)      { int d = pk - pkCompact[p - 1]; if ((float)d <= max_w && d < lw) lw = d; }
    if (p < np - 1) { int d = pkCompact[p + 1] - pk; if ((float)d <= max_w && d < rw) rw = d; }
    int li = imax2(pk - lw, 0);
    int ri = pk + rw;
    int ival = (ri - li) / 2;
    if (ival <= 0) { lv += imax2(period, 1); continue; }
    int a = lv - ival, b = lv + ival;
    int st = iclamp(a, 0, T), sp = iclamp(b, 0, T);
    int dst_len = imax2(sp - st, 0);
    int src_len = imax2(imin2(li + 2 * ival, T) - li, 0);
    int seglen = imin2(dst_len, src_len);
    float inv2i = 1.0f / (2.0f * (float)ival);
    for (int nIdx = threadIdx.x; nIdx < seglen; nIdx += blockDim.x) {
      float wv = 0.5f - 0.5f * __cosf(6.28318530718f * (float)nIdx * inv2i);
      atomicAdd(&out[st + nIdx], wv * (snd[li + nIdx] - mean));
    }
    lv += 2 * ival;
  }
}

// ---------------- kernel 8: finalize tail ----------------
__global__ void finalizeKernel(const float* __restrict__ snd, const float* __restrict__ meanVal,
                               const int* __restrict__ tailPtr, float* __restrict__ out, int T) {
  int tail = *tailPtr; float mean = *meanVal;
  for (long long t = (long long)blockIdx.x * blockDim.x + threadIdx.x; t < T;
       t += (long long)gridDim.x * blockDim.x)
    if (t >= tail) out[t] = snd[t] - mean;
}

// ---------------- host entry ----------------
extern "C" void kernel_launch(void* const* d_in, const int* in_sizes, int n_in,
                              void* d_out, int out_size, void* d_ws, size_t ws_size,
                              hipStream_t stream) {
  const float* snd    = (const float*)d_in[0];
  const float* pitch  = (const float*)d_in[1];
  const float* pshift = (const float*)d_in[2];
  const float* prange = (const float*)d_in[3];
  int T = in_sizes[0];
  int S = in_sizes[1];
  float* out = (float*)d_out;
  (void)n_in; (void)out_size; (void)ws_size;

  char* w = (char*)d_ws;
  size_t off = 0;
  auto take = [&](size_t bytes) -> void* {
    off = (off + 255) & ~(size_t)255;
    void* p = (void*)(w + off);
    off += bytes;
    return p;
  };
  float*    sumAccum     = (float*)take(4);
  float*    meanVal      = (float*)take(4);
  unsigned* gpeakBits    = (unsigned*)take(4);
  unsigned* f0minBits    = (unsigned*)take(4);
  float*    medianVal    = (float*)take(4);
  int*      tailPtr      = (int*)take(4);
  int*      cnts         = (int*)take(8);
  int*      nSegPtr      = (int*)take(4);
  int*      nPeaksPtr    = (int*)take(4);
  unsigned* hist         = (unsigned*)take((size_t)HBINS * 4);
  int*      segStartRaw  = (int*)take((size_t)MAXSEG * 4);
  int*      segEndRaw    = (int*)take((size_t)MAXSEG * 4);
  int*      segStart     = (int*)take((size_t)MAXSEG * 4);
  int*      segEnd       = (int*)take((size_t)MAXSEG * 4);
  int*      segPeakCount = (int*)take((size_t)MAXSEG * 4);
  int*      segPeakOff   = (int*)take((size_t)MAXSEG * 4);
  _Float16* xh           = (_Float16*)take((size_t)T * 2);
  int*      pkStage      = (int*)take((size_t)MAXSEG * PKSTRIDE * 4);
  int*      pkCompact    = (int*)take((size_t)MAXPK * 4);

  int tpb = 256;
  int gridT = (T + tpb - 1) / tpb;

  initKernel<<<gridT, tpb, 0, stream>>>(out, T, hist, HBINS, sumAccum, gpeakBits, f0minBits,
                                        cnts, nSegPtr, nPeaksPtr, segPeakCount, MAXSEG);
  sumKernel<<<2048, 256, 0, stream>>>(snd, sumAccum, T);
  meanKernel<<<1, 32, 0, stream>>>(sumAccum, meanVal, T);
  prepKernel<<<2048, 256, 0, stream>>>(snd, pitch, meanVal, xh, gpeakBits, f0minBits, hist,
                                       segStartRaw, segEndRaw, cnts, S, T);
  segSortKernel<<<1, 32, 0, stream>>>(cnts, segStartRaw, segEndRaw, segStart, segEnd,
                                      nSegPtr, tailPtr, hist, medianVal, pshift, T);
  findPeaksKernel<<<MAXSEG, 32, 0, stream>>>(snd, pitch, xh, meanVal, gpeakBits, nSegPtr,
                                             segStart, segEnd, pkStage, segPeakCount, S, T);
  compactKernel<<<1, 256, 0, stream>>>(nSegPtr, segPeakCount, segPeakOff, pkStage,
                                       pkCompact, nPeaksPtr);
  psolaKernel<<<MAXSEG, 128, 0, stream>>>(snd, pitch, meanVal, medianVal, f0minBits, pshift,
                                          prange, nSegPtr, segStart, segEnd, pkCompact,
                                          nPeaksPtr, out, S, T);
  finalizeKernel<<<gridT, tpb, 0, stream>>>(snd, meanVal, tailPtr, out, T);
}
